// MultiHeadAttention_72799695667391
// MI455X (gfx1250) — matrix operand order repaired and checked by reading, not compiled, not run on hardware
//
#include <hip/hip_runtime.h>

typedef __attribute__((ext_vector_type(16))) __bf16 v16bf;
typedef __attribute__((ext_vector_type(8)))  __bf16 v8bf;
typedef __attribute__((ext_vector_type(8)))  float  v8f;

union U8b { uint4 u; __bf16 b[8]; };
union U4b { uint2 u; __bf16 b[4]; };

__device__ __forceinline__ v8f wmma_bf16(v16bf a, v16bf b, v8f c) {
  // D = A(16x32) * B(32x16) + C, f32 accumulate
  return __builtin_amdgcn_wmma_f32_16x16x32_bf16(false, a, false, b, (short)0, c, false, false);
}

__device__ __forceinline__ v8f vzero8() {
  v8f z;
#pragma unroll
  for (int i = 0; i < 8; ++i) z[i] = 0.0f;
  return z;
}

// CDNA5 async global->LDS copy, 16B per lane, tracked with ASYNCcnt.
// lds_addr = low 32 bits of flat shared pointer (ISA: flat LDS addr = addr[31:0]).
__device__ __forceinline__ void async_ld16(void* lds_ptr, const void* gptr) {
  unsigned int lds_addr = (unsigned int)(uintptr_t)lds_ptr;
  unsigned long long ga = (unsigned long long)(uintptr_t)gptr;
  asm volatile("global_load_async_to_lds_b128 %0, %1, off"
               :: "v"(lds_addr), "v"(ga)
               : "memory");
}
__device__ __forceinline__ void wait_async_le4() {
  asm volatile("s_wait_asynccnt 0x4" ::: "memory");
}
__device__ __forceinline__ void wait_async_0() {
  asm volatile("s_wait_asynccnt 0x0" ::: "memory");
}

// A fragment (16x32 bf16). rowptr = &src[row][k0] with row = lane&15 (caller).
// Lanes 0-15: K = {0..7, 16..23}; lanes 16-31: K = {8..15, 24..31} (ISA 7.12.2).
__device__ __forceinline__ v16bf load_afrag(const __bf16* rowptr, int lane) {
  const int kb = (lane & 16) ? 8 : 0;
  v8bf lo = *(const v8bf*)(rowptr + kb);
  v8bf hi = *(const v8bf*)(rowptr + kb + 16);
  v16bf r;
#pragma unroll
  for (int i = 0; i < 8; ++i) { r[i] = lo[i]; r[i + 8] = hi[i]; }
  return r;
}

// B fragment (32x16 bf16) from K-major storage Bt[n][k] (B transposed).
// rowptr = &Bt[col][k0], col = lane&15 (caller). Lanes 0-15: K = {0..7, 16..23}; lanes 16-31: K = {8..15, 24..31}.
__device__ __forceinline__ v16bf load_bfrag(const __bf16* rowptr, int lane) {
  const int kb = (lane & 16) ? 8 : 0;
  v8bf lo = *(const v8bf*)(rowptr + kb);
  v8bf hi = *(const v8bf*)(rowptr + kb + 16);
  v16bf r;
#pragma unroll
  for (int i = 0; i < 8; ++i) { r[i] = lo[i]; r[i + 8] = hi[i]; }
  return r;
}

__global__ __launch_bounds__(256) void f32_to_bf16_k(const float* __restrict__ in,
                                                     __bf16* __restrict__ out, int n4) {
  int i = blockIdx.x * blockDim.x + threadIdx.x;
  if (i < n4) {
    float4 f = ((const float4*)in)[i];
    U4b t;
    t.b[0] = (__bf16)f.x; t.b[1] = (__bf16)f.y;
    t.b[2] = (__bf16)f.z; t.b[3] = (__bf16)f.w;
    ((uint2*)out)[i] = t.u;
  }
}

// Y[m,n] = sum_k A[m,k] * W[n,k]   (A: MxK row-major, W: NxK row-major -> "NT")
// If Yf != nullptr: fp32 output + bias. Else bf16 output to Ybf.
// Double-buffered LDS fed by CDNA5 async global->LDS DMA.
#define GT_M 128
#define GT_N 128
#define GT_K 32
__global__ __launch_bounds__(256) void gemm_nt_bf16(
    const __bf16* __restrict__ A, const __bf16* __restrict__ W,
    __bf16* __restrict__ Ybf, float* __restrict__ Yf, const float* __restrict__ bias,
    int M, int N, int K) {
  __shared__ __bf16 ldsA[2][GT_M][GT_K + 8];   // row = 80B, 16B aligned
  __shared__ __bf16 ldsW[2][GT_N][GT_K + 8];
  const int tid = threadIdx.x;
  const int lane = tid & 31, wave = tid >> 5;
  const int wm = wave >> 1, wn = wave & 1;          // 4x2 wave grid
  const int m0 = blockIdx.y * GT_M, n0 = blockIdx.x * GT_N;
  const int cl = lane & 15;
  const int rsel = (lane & 16) ? 8 : 0;

  // per-thread copy chunks: tile = 128 rows x 4 chunks(8 elts) = 512; 2 per thread
  const int r0 = tid >> 2,          c0 = (tid & 3) << 3;
  const int r1 = (tid + 256) >> 2,  c1 = ((tid + 256) & 3) << 3;

  auto issue = [&](int buf, int k0) {
    async_ld16(&ldsA[buf][r0][c0], &A[(size_t)(m0 + r0) * K + k0 + c0]);
    async_ld16(&ldsA[buf][r1][c1], &A[(size_t)(m0 + r1) * K + k0 + c1]);
    async_ld16(&ldsW[buf][r0][c0], &W[(size_t)(n0 + r0) * K + k0 + c0]);
    async_ld16(&ldsW[buf][r1][c1], &W[(size_t)(n0 + r1) * K + k0 + c1]);
  };

  v8f acc[2][4];
#pragma unroll
  for (int i = 0; i < 2; ++i)
#pragma unroll
    for (int j = 0; j < 4; ++j) acc[i][j] = vzero8();

  issue(0, 0);
  const int nk = K / GT_K;
  for (int kt = 0; kt < nk; ++kt) {
    const int buf = kt & 1;
    if (kt + 1 < nk) {
      issue(buf ^ 1, (kt + 1) * GT_K);
      wait_async_le4();      // current tile's 4 copies done; next 4 in flight
    } else {
      wait_async_0();
    }
    __syncthreads();

    v16bf af[2], bf[4];
#pragma unroll
    for (int i = 0; i < 2; ++i)
      af[i] = load_afrag(&ldsA[buf][wm * 32 + i * 16 + cl][0], lane);
#pragma unroll
    for (int j = 0; j < 4; ++j)
      bf[j] = load_bfrag(&ldsW[buf][wn * 64 + j * 16 + cl][0], lane);
#pragma unroll
    for (int i = 0; i < 2; ++i)
#pragma unroll
      for (int j = 0; j < 4; ++j)
        acc[i][j] = wmma_bf16(af[i], bf[j], acc[i][j]);

    __syncthreads();         // readers done before next issue overwrites buffers
  }

  // C/D layout: lane -> col = cl, rows = rsel + v (ISA 7.12.2)
#pragma unroll
  for (int i = 0; i < 2; ++i) {
#pragma unroll
    for (int j = 0; j < 4; ++j) {
      int col = n0 + wn * 64 + j * 16 + cl;
#pragma unroll
      for (int v = 0; v < 8; ++v) {
        int row = m0 + wm * 32 + i * 16 + rsel + v;
        float val = acc[i][j][v];
        if (Yf) {
          Yf[(size_t)row * N + col] = val + bias[col];
        } else {
          Ybf[(size_t)row * N + col] = (__bf16)val;
        }
      }
    }
  }
}

// Flash attention: one block per (b, h, 128-query tile). 8 waves, wave w owns
// query rows 16w..16w+15 of the tile. HD = 64.
#define AQ 128
#define AK 128
#define AHD 64
__global__ __launch_bounds__(256) void flash_attn_bf16(
    const __bf16* __restrict__ Q, const __bf16* __restrict__ Km,
    const __bf16* __restrict__ V, __bf16* __restrict__ O,
    int S, int Dm, int H) {
  __shared__ __bf16 ldsQ[AQ][AHD + 8];          // 144B rows
  __shared__ __bf16 ldsK[AK][AHD + 8];
  __shared__ __bf16 ldsVt[AHD][AK + 8];         // V transposed: [d][j], 272B rows
  __shared__ __bf16 ldsP[8][16][AK + 8];        // wave-private P tiles

  const int tid = threadIdx.x;
  const int lane = tid & 31, wave = tid >> 5;
  const int cl = lane & 15;
  const int rsel = (lane & 16) ? 8 : 0;
  const int qb = blockIdx.x;
  const int bh = blockIdx.y;
  const int b = bh / H, h = bh % H;
  const int q0 = qb * AQ;
  const size_t base = ((size_t)b * S) * Dm + (size_t)h * AHD;
  const float scale = 0.125f;                    // 1/sqrt(64)
  const float NEG = -1.0e30f;

  // Load Q tile (128x64) via async DMA: 1024 8-elt chunks, 4 per thread
#pragma unroll
  for (int i = 0; i < 4; ++i) {
    int idx = tid + i * 256;
    int r = idx >> 3, c = (idx & 7) << 3;
    async_ld16(&ldsQ[r][c], &Q[base + (size_t)(q0 + r) * Dm + c]);
  }
  wait_async_0();
  __syncthreads();

  v16bf aq0 = load_afrag(&ldsQ[wave * 16 + cl][0], lane);
  v16bf aq1 = load_afrag(&ldsQ[wave * 16 + cl][32], lane);

  float mrow[8], lrow[8];
  v8f ctx[4];
#pragma unroll
  for (int v = 0; v < 8; ++v) { mrow[v] = NEG; lrow[v] = 0.0f; }
#pragma unroll
  for (int t = 0; t < 4; ++t) ctx[t] = vzero8();

  for (int kb = 0; kb <= qb; ++kb) {
    const int k0 = kb * AK;
    __syncthreads();   // previous iteration's ldsK/ldsVt reads done
    // K tile (128x64) via async DMA
#pragma unroll
    for (int i = 0; i < 4; ++i) {
      int idx = tid + i * 256;
      int r = idx >> 3, c = (idx & 7) << 3;
      async_ld16(&ldsK[r][c], &Km[base + (size_t)(k0 + r) * Dm + c]);
    }
    // V tile transposed into ldsVt[d][j] (needs VGPR round-trip for transpose)
#pragma unroll
    for (int i = 0; i < 4; ++i) {
      int idx = tid + i * 256;
      int r = idx >> 3, c = (idx & 7) << 3;
      U8b t;
      t.u = *(const uint4*)&V[base + (size_t)(k0 + r) * Dm + c];
#pragma unroll
      for (int e = 0; e < 8; ++e) ldsVt[c + e][r] = t.b[e];
    }
    wait_async_0();
    __syncthreads();

    // Scores S = Q K^T : wave computes 16x128 (8 n-tiles, 2 k-steps of 32)
    v8f sc[8];
#pragma unroll
    for (int nt = 0; nt < 8; ++nt) {
      v16bf bk0 = load_bfrag(&ldsK[nt * 16 + cl][0], lane);
      v16bf bk1 = load_bfrag(&ldsK[nt * 16 + cl][32], lane);
      v8f t = wmma_bf16(aq0, bk0, vzero8());
      sc[nt] = wmma_bf16(aq1, bk1, t);
    }

    // scale + causal mask (only the diagonal block needs masking)
#pragma unroll
    for (int nt = 0; nt < 8; ++nt) {
#pragma unroll
      for (int v = 0; v < 8; ++v) {
        float x = sc[nt][v] * scale;
        if (kb == qb) {
          int kc = k0 + nt * 16 + cl;
          int qr = q0 + wave * 16 + rsel + v;
          if (kc > qr) x = NEG;
        }
        sc[nt][v] = x;
      }
    }

    // online softmax update (row = (rsel+v), cols spread across 16-lane group)
    float al[8];
#pragma unroll
    for (int v = 0; v < 8; ++v) {
      float rm = sc[0][v];
#pragma unroll
      for (int nt = 1; nt < 8; ++nt) rm = fmaxf(rm, sc[nt][v]);
      rm = fmaxf(rm, __shfl_xor(rm, 8, 32));
      rm = fmaxf(rm, __shfl_xor(rm, 4, 32));
      rm = fmaxf(rm, __shfl_xor(rm, 2, 32));
      rm = fmaxf(rm, __shfl_xor(rm, 1, 32));
      float mn = fmaxf(mrow[v], rm);
      al[v] = __expf(mrow[v] - mn);
      mrow[v] = mn;
    }
#pragma unroll
    for (int nt = 0; nt < 8; ++nt)
#pragma unroll
      for (int v = 0; v < 8; ++v)
        sc[nt][v] = __expf(sc[nt][v] - mrow[v]);
#pragma unroll
    for (int v = 0; v < 8; ++v) {
      float rs = 0.0f;
#pragma unroll
      for (int nt = 0; nt < 8; ++nt) rs += sc[nt][v];
      rs += __shfl_xor(rs, 8, 32);
      rs += __shfl_xor(rs, 4, 32);
      rs += __shfl_xor(rs, 2, 32);
      rs += __shfl_xor(rs, 1, 32);
      lrow[v] = lrow[v] * al[v] + rs;
    }
#pragma unroll
    for (int t = 0; t < 4; ++t)
#pragma unroll
      for (int v = 0; v < 8; ++v) ctx[t][v] *= al[v];

    // P (bf16) to wave-private LDS, then ctx += P @ V
#pragma unroll
    for (int nt = 0; nt < 8; ++nt)
#pragma unroll
      for (int v = 0; v < 8; ++v)
        ldsP[wave][rsel + v][nt * 16 + cl] = (__bf16)sc[nt][v];
    __syncthreads();

#pragma unroll
    for (int ks = 0; ks < 4; ++ks) {
      v16bf ap = load_afrag(&ldsP[wave][cl][ks * 32], lane);
#pragma unroll
      for (int dt = 0; dt < 4; ++dt) {
        v16bf bv = load_bfrag(&ldsVt[dt * 16 + cl][ks * 32], lane);
        ctx[dt] = wmma_bf16(ap, bv, ctx[dt]);
      }
    }
  }

  // normalize and write ctx (bf16) back as [b, s, h*64 + d]
#pragma unroll
  for (int dt = 0; dt < 4; ++dt) {
#pragma unroll
    for (int v = 0; v < 8; ++v) {
      int qr = q0 + wave * 16 + rsel + v;
      int col = h * AHD + dt * 16 + cl;
      O[((size_t)b * S + qr) * Dm + col] = (__bf16)(ctx[dt][v] / lrow[v]);
    }
  }
}

extern "C" void kernel_launch(void* const* d_in, const int* in_sizes, int n_in,
                              void* d_out, int out_size, void* d_ws, size_t ws_size,
                              hipStream_t stream) {
  (void)in_sizes; (void)n_in; (void)out_size; (void)ws_size;
  const int B = 4, S = 2048, D = 1024, H = 16;
  const int M = B * S;                 // 8192

  const float* x  = (const float*)d_in[0];
  const float* Wq = (const float*)d_in[1];
  const float* Wk = (const float*)d_in[2];
  const float* Wv = (const float*)d_in[3];
  const float* Wo = (const float*)d_in[4];
  const float* bo = (const float*)d_in[5];
  float* out = (float*)d_out;

  // carve workspace (256B aligned slots)
  uintptr_t p = (uintptr_t)d_ws;
  auto carve = [&](size_t bytes) {
    uintptr_t r = p;
    p += (bytes + 255) & ~(size_t)255;
    return (__bf16*)r;
  };
  const size_t xd = (size_t)M * D, wd = (size_t)D * D;
  __bf16* xb  = carve(xd * 2);
  __bf16* wqb = carve(wd * 2);
  __bf16* wkb = carve(wd * 2);
  __bf16* wvb = carve(wd * 2);
  __bf16* wob = carve(wd * 2);
  __bf16* Qb  = carve(xd * 2);
  __bf16* Kb  = carve(xd * 2);
  __bf16* Vb  = carve(xd * 2);
  __bf16* Cb  = carve(xd * 2);

  // fp32 -> bf16 conversions
  {
    int n4 = (int)(xd / 4);
    f32_to_bf16_k<<<n4 / 256, 256, 0, stream>>>(x, xb, n4);
    int w4 = (int)(wd / 4);
    f32_to_bf16_k<<<w4 / 256, 256, 0, stream>>>(Wq, wqb, w4);
    f32_to_bf16_k<<<w4 / 256, 256, 0, stream>>>(Wk, wkb, w4);
    f32_to_bf16_k<<<w4 / 256, 256, 0, stream>>>(Wv, wvb, w4);
    f32_to_bf16_k<<<w4 / 256, 256, 0, stream>>>(Wo, wob, w4);
  }

  // QKV projections: Y = x @ W^T
  dim3 gg(D / GT_N, M / GT_M);     // (8, 64)
  gemm_nt_bf16<<<gg, 256, 0, stream>>>(xb, wqb, Qb, nullptr, nullptr, M, D, D);
  gemm_nt_bf16<<<gg, 256, 0, stream>>>(xb, wkb, Kb, nullptr, nullptr, M, D, D);
  gemm_nt_bf16<<<gg, 256, 0, stream>>>(xb, wvb, Vb, nullptr, nullptr, M, D, D);

  // causal flash attention
  dim3 ga(S / AQ, B * H);          // (16, 64)
  flash_attn_bf16<<<ga, 256, 0, stream>>>(Qb, Kb, Vb, Cb, S, D, H);

  // output projection + bias, fp32 out
  gemm_nt_bf16<<<gg, 256, 0, stream>>>(Cb, wob, nullptr, out, bo, M, D, D);
}
